// MNT_84224308674640
// MI455X (gfx1250) — matrix operand order repaired and checked by reading, not compiled, not run on hardware
//
#include <hip/hip_runtime.h>
#include <hip/hip_bf16.h>
#include <math.h>

// ---------------------------------------------------------------------------
// CDNA5 (gfx1250) ViT block: patch-embed + LN + PE, FFN, windowed attention,
// FFN, residuals.  GEMMs: bf16 WMMA (v_wmma_f32_16x16x32_bf16, f32 acc).
// Tile staging: Tensor Data Mover (tensor_load_to_lds + s_wait_tensorcnt).
// ---------------------------------------------------------------------------

typedef __attribute__((ext_vector_type(16))) __bf16        v16bf;
typedef __attribute__((ext_vector_type(8)))  float         v8f;
typedef __attribute__((ext_vector_type(4)))  unsigned int  v4u;
typedef __attribute__((ext_vector_type(8)))  int           v8i;
typedef __attribute__((ext_vector_type(4)))  int           v4i;

union FragB16 {
    v16bf v;
    uint4 q[2];
};

__device__ __forceinline__ unsigned short f2b(float f) {
    union { float f; unsigned int u; } x;
    x.f = f;
    unsigned int u = x.u;
    unsigned int r = u + 0x7fffu + ((u >> 16) & 1u);   // round-to-nearest-even
    return (unsigned short)(r >> 16);
}

__device__ __forceinline__ float gelu_tanh(float x) {
    float x3 = x * x * x;
    return 0.5f * x * (1.0f + tanhf(0.7978845608028654f * (x + 0.044715f * x3)));
}

// LDS byte offset of a __shared__ object (generic ptr low 32 bits = LDS offset)
__device__ __forceinline__ unsigned lds_off(const void* p) {
    return (unsigned)(uintptr_t)p;
}

// ---------------------------------------------------------------------------
// Tensor Data Mover: async DMA of a 2D bf16 tile (tileD0 x tileD1 elements,
// row stride strideElems) from global memory into LDS at ldsByteAddr.
// D# packed per CDNA5 ISA ch.8 (group0: count/lds/global/type, group1: dims).
// ---------------------------------------------------------------------------
__device__ __forceinline__ void tdm_load_2d_bf16(unsigned ldsByteAddr,
                                                 const void* gptr,
                                                 unsigned long long strideElems,
                                                 unsigned tensorD0,
                                                 unsigned tensorD1,
                                                 unsigned tileD0,
                                                 unsigned tileD1) {
    unsigned long long ga = (unsigned long long)(uintptr_t)gptr;

    v4u g0;
    g0[0] = 1u;                                            // count=1, user D#
    g0[1] = ldsByteAddr;                                   // lds_addr
    g0[2] = (unsigned)(ga & 0xffffffffull);                // global_addr[31:0]
    g0[3] = (unsigned)((ga >> 32) & 0x1ffffffull)          // global_addr[56:32]
          | (2u << 30);                                    // type = 2 ("image")

    v8i g1;
    g1[0] = (int)(1u << 16);                               // data_size=1 (2B)
    g1[1] = (int)((tensorD0 & 0xffffu) << 16);             // tensor_dim0 lo
    g1[2] = (int)((tensorD0 >> 16) | ((tensorD1 & 0xffffu) << 16));
    g1[3] = (int)((tensorD1 >> 16) | (tileD0 << 16));      // tile_dim0
    g1[4] = (int)(tileD1 & 0xffffu);                       // tile_dim1, dim2=0
    g1[5] = (int)(strideElems & 0xffffffffull);            // dim0_stride lo
    g1[6] = (int)((strideElems >> 32) & 0xffffull);        // dim0_stride hi
    g1[7] = 0;                                             // dim1_stride = 0

    v4i z4 = {};
#if __has_include(<hip/amd_detail/amd_gfx1250_TDM.h>)
    // amdgpu-toolchain (clang-23 / therock-10.0): 6-arg builtin
    v8i z8 = {};
    __builtin_amdgcn_tensor_load_to_lds(g0, g1, z4, z4, z8, 0);
#else
    // ROCm 7.2 (clang-22): 5-arg builtin
    __builtin_amdgcn_tensor_load_to_lds(g0, g1, z4, z4, 0);
#endif
}

// -------------------------- prep kernels -----------------------------------

__global__ void k_convert_bf16(const float* __restrict__ src,
                               unsigned short* __restrict__ dst, int n) {
    int i = blockIdx.x * blockDim.x + threadIdx.x;
    if (i < n) dst[i] = f2b(src[i]);
}

// src: K x N (row-major f32)  ->  dst: N x K (row-major bf16)
__global__ void k_transpose_bf16(const float* __restrict__ src,
                                 unsigned short* __restrict__ dst,
                                 int K, int N) {
    int i = blockIdx.x * blockDim.x + threadIdx.x;
    if (i >= K * N) return;
    int k = i / N;
    int n = i - k * N;
    dst[(size_t)n * K + k] = f2b(src[i]);
}

// input (16,3,512,512) -> Aim (16384 x 768) bf16, row=(b,ph,pw), col=(c,py,px)
__global__ void k_im2col(const float* __restrict__ in,
                         unsigned short* __restrict__ dst) {
    int i = blockIdx.x * blockDim.x + threadIdx.x;   // < 16384*768
    int row = i / 768;
    int k = i - row * 768;
    int b  = row >> 10;          // /1024
    int p  = row & 1023;
    int ph = p >> 5;             // /32
    int pw = p & 31;
    int c  = k >> 8;             // /256
    int py = (k >> 4) & 15;
    int px = k & 15;
    size_t src = (((size_t)(b * 3 + c) * 512) + (size_t)(ph * 16 + py)) * 512
               + (size_t)(pw * 16 + px);
    dst[(size_t)row * 768 + k] = f2b(in[src]);
}

// -------------------------- generic WMMA GEMM ------------------------------
// C[M,N] = A[M,K](bf16) * WT[N,K]^T(bf16) + bias, optional gelu/residual.
// Block tile 128x64, BK=32, 256 threads = 8 waves (4 along M x 2 along N),
// each wave computes 32x32 via 4 x v_wmma_f32_16x16x32_bf16 per K step.
// LDS tiles double-buffered, filled by the Tensor Data Mover (wave 0).

enum { GF_GELU = 1, GF_RES = 2, GF_WF32 = 4, GF_WB16 = 8 };

__global__ __launch_bounds__(256)
void k_gemm_bf16(const unsigned short* __restrict__ A,
                 const unsigned short* __restrict__ WT,
                 const float* __restrict__ bias,
                 const float* __restrict__ resid,
                 float* __restrict__ outF,
                 unsigned short* __restrict__ outB,
                 int M, int N, int K, int flags) {
    __shared__ unsigned short As[2][128 * 32];   // [m][k], stride 32
    __shared__ unsigned short Bs[2][64 * 32];    // [n][k], stride 32

    const int tid  = threadIdx.x;
    const int lane = tid & 31;
    const int wave = tid >> 5;
    const int wm   = wave & 3;      // 0..3  -> 32-row slab
    const int wn   = wave >> 2;     // 0..1  -> 32-col slab
    const int half = lane >> 4;
    const int ln   = lane & 15;

    const int blockRow = blockIdx.y * 128;
    const int blockCol = blockIdx.x * 64;
    const int nT = K >> 5;          // number of 32-wide K tiles

    v8f acc[2][2] = {};

    // prime buffer 0 via TDM (wave 0 only; EXEC-independent DMA)
    if (wave == 0) {
        tdm_load_2d_bf16(lds_off(&As[0][0]),
                         A + (size_t)blockRow * K, (unsigned long long)K,
                         (unsigned)K, (unsigned)M, 32u, 128u);
        tdm_load_2d_bf16(lds_off(&Bs[0][0]),
                         WT + (size_t)blockCol * K, (unsigned long long)K,
                         (unsigned)K, (unsigned)N, 32u, 64u);
    }

    for (int t = 0; t < nT; ++t) {
        // all waves done reading buf[(t+1)&1] (from iteration t-1)
        __syncthreads();
        if (wave == 0) {
            if (t + 1 < nT) {
                int k0 = (t + 1) << 5;
                tdm_load_2d_bf16(lds_off(&As[(t + 1) & 1][0]),
                                 A + (size_t)blockRow * K + k0,
                                 (unsigned long long)K,
                                 (unsigned)K, (unsigned)M, 32u, 128u);
                tdm_load_2d_bf16(lds_off(&Bs[(t + 1) & 1][0]),
                                 WT + (size_t)blockCol * K + k0,
                                 (unsigned long long)K,
                                 (unsigned)N * 0 + (unsigned)K, (unsigned)N,
                                 32u, 64u);
                // two newest ops may stay in flight; current tiles complete
                __builtin_amdgcn_s_wait_tensorcnt((short)2);
            } else {
                __builtin_amdgcn_s_wait_tensorcnt((short)0);
            }
        } else if (t + 1 < nT) {
            // warm GL2 for the next DMA tiles (global_prefetch_b8)
            int k0 = (t + 1) << 5;
            __builtin_prefetch(A + (size_t)(blockRow + (tid & 127)) * K + k0, 0, 1);
            __builtin_prefetch(WT + (size_t)(blockCol + (tid & 63)) * K + k0, 0, 1);
        }
        // data-ready barrier for buf[t&1]
        __syncthreads();

        const unsigned short* Ab = As[t & 1];
        const unsigned short* Bb = Bs[t & 1];

        FragB16 a[2], b[2];
#pragma unroll
        for (int i = 0; i < 2; ++i) {
            int m = wm * 32 + i * 16 + ln;
            a[i].q[0] = *(const uint4*)&Ab[m * 32 + half * 8];
            a[i].q[1] = *(const uint4*)&Ab[m * 32 + 16 + half * 8];
        }
#pragma unroll
        for (int j = 0; j < 2; ++j) {
            int n = wn * 32 + j * 16 + ln;
            b[j].q[0] = *(const uint4*)&Bb[n * 32 + half * 8];
            b[j].q[1] = *(const uint4*)&Bb[n * 32 + 16 + half * 8];
        }
#pragma unroll
        for (int i = 0; i < 2; ++i)
#pragma unroll
            for (int j = 0; j < 2; ++j)
                acc[i][j] = __builtin_amdgcn_wmma_f32_16x16x32_bf16(
                    false, a[i].v, false, b[j].v, (short)0, acc[i][j],
                    false, false);
    }

    // epilogue
#pragma unroll
    for (int i = 0; i < 2; ++i) {
#pragma unroll
        for (int j = 0; j < 2; ++j) {
            int n = blockCol + wn * 32 + j * 16 + ln;
            float bv = bias[n];
#pragma unroll
            for (int r = 0; r < 8; ++r) {
                int m = blockRow + wm * 32 + i * 16 + r + 8 * half;
                float v = acc[i][j][r] + bv;
                if (flags & GF_GELU) v = gelu_tanh(v);
                if (flags & GF_RES)  v += resid[(size_t)m * N + n];
                if (flags & GF_WF32) outF[(size_t)m * N + n] = v;
                if (flags & GF_WB16) outB[(size_t)m * N + n] = f2b(v);
            }
        }
    }
}

// -------------------------- LayerNorm + sinusoidal PE ----------------------
// one block (256 threads) per row of 768

__global__ __launch_bounds__(256)
void k_ln_pe(const float* __restrict__ in,
             const float* __restrict__ g,
             const float* __restrict__ b,
             float* __restrict__ outF,
             unsigned short* __restrict__ outB) {
    __shared__ float red[16];
    const int row  = blockIdx.x;
    const int tid  = threadIdx.x;
    const int lane = tid & 31;
    const int wave = tid >> 5;
    const float* x = in + (size_t)row * 768;

    float v0 = x[tid], v1 = x[tid + 256], v2 = x[tid + 512];
    float s  = v0 + v1 + v2;
    float ss = v0 * v0 + v1 * v1 + v2 * v2;
#pragma unroll
    for (int off = 1; off < 32; off <<= 1) {
        s  += __shfl_xor(s,  off, 32);
        ss += __shfl_xor(ss, off, 32);
    }
    if (lane == 0) { red[wave] = s; red[8 + wave] = ss; }
    __syncthreads();
    float ts = 0.f, tss = 0.f;
#pragma unroll
    for (int w = 0; w < 8; ++w) { ts += red[w]; tss += red[8 + w]; }
    float mean = ts * (1.0f / 768.0f);
    float var  = tss * (1.0f / 768.0f) - mean * mean;
    float rstd = rsqrtf(var + 1e-5f);

    float pos = (float)(row & 1023);     // token index within sequence
#pragma unroll
    for (int e = 0; e < 3; ++e) {
        int i = tid + e * 256;
        float xv = (e == 0) ? v0 : (e == 1) ? v1 : v2;
        float y  = (xv - mean) * rstd * g[i] + b[i];
        float ang = pos * __powf(10000.0f, -(float)(i & ~1) / 768.0f);
        y += (i & 1) ? __cosf(ang) : __sinf(ang);
        outF[(size_t)row * 768 + i] = y;
        outB[(size_t)row * 768 + i] = f2b(y);
    }
}

// -------------------------- windowed attention -----------------------------
// grid: (B*nW, NH); block: 128 threads (4 waves); per block one 64x64 window
// for one head.  scores = QK^T * 1/8, softmax, O = P V.  All matmuls WMMA.
// Q/K tiles staged by the TDM; V staged (transposed) by VALU loads in
// parallel with the DMA.

__global__ __launch_bounds__(128)
void k_attn(const unsigned short* __restrict__ qkv,   // [16384, 2304] bf16
            unsigned short* __restrict__ obuf) {      // [16384, 768]  bf16
    __shared__ unsigned short Qs[64 * 64];
    __shared__ unsigned short Ks[64 * 64];
    __shared__ unsigned short VsT[64 * 64];   // [d][token]
    __shared__ unsigned short Ps[64 * 64];

    const int tid  = threadIdx.x;
    const int lane = tid & 31;
    const int wave = tid >> 5;           // 0..3
    const int half = lane >> 4;
    const int ln   = lane & 15;

    const int bw = blockIdx.x;           // b*16 + w
    const int b  = bw >> 4;
    const int w  = bw & 15;
    const int hh = blockIdx.y;
    const size_t rowBase = (size_t)(b * 1024 + w * 64);

    // TDM: Q (64x64) and K (64x64) window tiles, row stride 2304
    if (wave == 0) {
        tdm_load_2d_bf16(lds_off(&Qs[0]),
                         qkv + rowBase * 2304 + hh * 64,
                         2304ull, 2304u, 16384u, 64u, 64u);
        tdm_load_2d_bf16(lds_off(&Ks[0]),
                         qkv + rowBase * 2304 + 768 + hh * 64,
                         2304ull, 2304u, 16384u, 64u, 64u);
    }
    // stage V transposed (VsT[d][token]) while the DMA runs
    for (int idx = tid; idx < 4096; idx += 128) {
        int tt = idx >> 6, dd = idx & 63;
        VsT[dd * 64 + tt] =
            qkv[(rowBase + tt) * 2304 + 1536 + hh * 64 + dd];
    }
    if (wave == 0) __builtin_amdgcn_s_wait_tensorcnt((short)0);
    __syncthreads();

    const int m0 = wave * 16;

    // scores: S = Q K^T  (B-frag read from Ks rows = K^T automatically)
    v8f s[4] = {};
#pragma unroll
    for (int kk = 0; kk < 64; kk += 32) {
        FragB16 a;
        int m = m0 + ln;
        a.q[0] = *(const uint4*)&Qs[m * 64 + kk + half * 8];
        a.q[1] = *(const uint4*)&Qs[m * 64 + kk + 16 + half * 8];
#pragma unroll
        for (int j = 0; j < 4; ++j) {
            FragB16 bb;
            int n = j * 16 + ln;
            bb.q[0] = *(const uint4*)&Ks[n * 64 + kk + half * 8];
            bb.q[1] = *(const uint4*)&Ks[n * 64 + kk + 16 + half * 8];
            s[j] = __builtin_amdgcn_wmma_f32_16x16x32_bf16(
                false, a.v, false, bb.v, (short)0, s[j], false, false);
        }
    }

    // softmax rows (each accumulator register r = one row across a 16-lane half)
    const float scale = 0.125f;          // dh^-0.5, dh = 64
#pragma unroll
    for (int j = 0; j < 4; ++j)
#pragma unroll
        for (int r = 0; r < 8; ++r) s[j][r] *= scale;

#pragma unroll
    for (int r = 0; r < 8; ++r) {
        float mx = -1e30f;
#pragma unroll
        for (int j = 0; j < 4; ++j) mx = fmaxf(mx, s[j][r]);
#pragma unroll
        for (int off = 1; off < 16; off <<= 1)
            mx = fmaxf(mx, __shfl_xor(mx, off, 32));
        float e[4], sum = 0.f;
#pragma unroll
        for (int j = 0; j < 4; ++j) { e[j] = __expf(s[j][r] - mx); sum += e[j]; }
#pragma unroll
        for (int off = 1; off < 16; off <<= 1)
            sum += __shfl_xor(sum, off, 32);
        float inv = 1.0f / sum;
        int m = m0 + r + 8 * half;
#pragma unroll
        for (int j = 0; j < 4; ++j)
            Ps[m * 64 + j * 16 + ln] = f2b(e[j] * inv);
    }
    __syncthreads();

    // O = P V   (B-frag from VsT rows = V in K-major)
    v8f o[4] = {};
#pragma unroll
    for (int kk = 0; kk < 64; kk += 32) {
        FragB16 a;
        int m = m0 + ln;
        a.q[0] = *(const uint4*)&Ps[m * 64 + kk + half * 8];
        a.q[1] = *(const uint4*)&Ps[m * 64 + kk + 16 + half * 8];
#pragma unroll
        for (int j = 0; j < 4; ++j) {
            FragB16 bb;
            int n = j * 16 + ln;
            bb.q[0] = *(const uint4*)&VsT[n * 64 + kk + half * 8];
            bb.q[1] = *(const uint4*)&VsT[n * 64 + kk + 16 + half * 8];
            o[j] = __builtin_amdgcn_wmma_f32_16x16x32_bf16(
                false, a.v, false, bb.v, (short)0, o[j], false, false);
        }
    }

#pragma unroll
    for (int j = 0; j < 4; ++j)
#pragma unroll
        for (int r = 0; r < 8; ++r) {
            int m = m0 + r + 8 * half;
            int n = j * 16 + ln;
            obuf[(rowBase + m) * 768 + hh * 64 + n] = f2b(o[j][r]);
        }
}

// -------------------------- host orchestration -----------------------------

extern "C" void kernel_launch(void* const* d_in, const int* in_sizes, int n_in,
                              void* d_out, int out_size, void* d_ws, size_t ws_size,
                              hipStream_t stream) {
    (void)in_sizes; (void)n_in; (void)out_size; (void)ws_size;

    const float* input   = (const float*)d_in[0];
    const float* patch_w = (const float*)d_in[1];
    const float* patch_b = (const float*)d_in[2];
    const float* ln_g    = (const float*)d_in[3];
    const float* ln_b    = (const float*)d_in[4];
    const float* qkv_w   = (const float*)d_in[5];
    const float* qkv_b   = (const float*)d_in[6];
    const float* proj_w  = (const float*)d_in[7];
    const float* proj_b  = (const float*)d_in[8];
    const float* ff1_w   = (const float*)d_in[9];
    const float* ff1_b   = (const float*)d_in[10];
    const float* ff2_w   = (const float*)d_in[11];
    const float* ff2_b   = (const float*)d_in[12];

    const int D = 768, F = 3072, T3 = 2304, M = 16384;

    char* ws = (char*)d_ws;
    size_t off = 0;
    auto alloc = [&](size_t bytes) {
        size_t o = off;
        off = (off + bytes + 255) & ~(size_t)255;
        return o;
    };

    unsigned short* wPatch = (unsigned short*)(ws + alloc((size_t)D * D * 2));
    unsigned short* wQkvT  = (unsigned short*)(ws + alloc((size_t)T3 * D * 2));
    unsigned short* wFf1T  = (unsigned short*)(ws + alloc((size_t)F * D * 2));
    unsigned short* wFf2T  = (unsigned short*)(ws + alloc((size_t)D * F * 2));
    unsigned short* wProjT = (unsigned short*)(ws + alloc((size_t)D * D * 2));
    unsigned short* Aim    = (unsigned short*)(ws + alloc((size_t)M * D * 2));
    float*          tmp    = (float*)         (ws + alloc((size_t)M * D * 4));
    float*          x0f    = (float*)         (ws + alloc((size_t)M * D * 4));
    unsigned short* x0b    = (unsigned short*)(ws + alloc((size_t)M * D * 2));
    unsigned short* h      = (unsigned short*)(ws + alloc((size_t)M * F * 2));
    unsigned short* y      = (unsigned short*)(ws + alloc((size_t)M * D * 2));
    unsigned short* qkvb   = (unsigned short*)(ws + alloc((size_t)M * T3 * 2));
    unsigned short* obuf   = (unsigned short*)(ws + alloc((size_t)M * D * 2));
    float*          x1f    = (float*)         (ws + alloc((size_t)M * D * 4));
    unsigned short* x1b    = (unsigned short*)(ws + alloc((size_t)M * D * 2));

    // ---- weight conversion ----
    // patch_w (D, C*P*P) is already (N=d) x (K=c,p,q): straight convert.
    k_convert_bf16<<<(D * D + 255) / 256, 256, 0, stream>>>(patch_w, wPatch, D * D);
    k_transpose_bf16<<<(D * T3 + 255) / 256, 256, 0, stream>>>(qkv_w, wQkvT, D, T3);
    k_transpose_bf16<<<(D * F + 255) / 256, 256, 0, stream>>>(ff1_w, wFf1T, D, F);
    k_transpose_bf16<<<(F * D + 255) / 256, 256, 0, stream>>>(ff2_w, wFf2T, F, D);
    k_transpose_bf16<<<(D * D + 255) / 256, 256, 0, stream>>>(proj_w, wProjT, D, D);

    // ---- im2col ----
    k_im2col<<<(M * D + 255) / 256, 256, 0, stream>>>(input, Aim);

    // ---- patch embedding GEMM -> tmp (f32) ----
    k_gemm_bf16<<<dim3(D / 64, M / 128), 256, 0, stream>>>(
        Aim, wPatch, patch_b, nullptr, tmp, nullptr, M, D, D, GF_WF32);

    // ---- LN + positional encoding -> x0f / x0b ----
    k_ln_pe<<<M, 256, 0, stream>>>(tmp, ln_g, ln_b, x0f, x0b);

    // ---- FFN #1:  h = gelu(x0 @ ff1 + b) ----
    k_gemm_bf16<<<dim3(F / 64, M / 128), 256, 0, stream>>>(
        x0b, wFf1T, ff1_b, nullptr, nullptr, h, M, F, D, GF_GELU | GF_WB16);
    // y = h @ ff2 + b
    k_gemm_bf16<<<dim3(D / 64, M / 128), 256, 0, stream>>>(
        h, wFf2T, ff2_b, nullptr, nullptr, y, M, D, F, GF_WB16);

    // ---- QKV projection ----
    k_gemm_bf16<<<dim3(T3 / 64, M / 128), 256, 0, stream>>>(
        y, wQkvT, qkv_b, nullptr, nullptr, qkvb, M, T3, D, GF_WB16);

    // ---- windowed attention (16 windows/seq, 12 heads) ----
    k_attn<<<dim3(16 * 16, 12), 128, 0, stream>>>(qkvb, obuf);

    // ---- output projection + residual(x0) -> x1f / x1b ----
    k_gemm_bf16<<<dim3(D / 64, M / 128), 256, 0, stream>>>(
        obuf, wProjT, proj_b, x0f, x1f, x1b, M, D, D,
        GF_RES | GF_WF32 | GF_WB16);

    // ---- FFN #2 + residual(x1) -> d_out (f32) ----
    k_gemm_bf16<<<dim3(F / 64, M / 128), 256, 0, stream>>>(
        x1b, wFf1T, ff1_b, nullptr, nullptr, h, M, F, D, GF_GELU | GF_WB16);
    k_gemm_bf16<<<dim3(D / 64, M / 128), 256, 0, stream>>>(
        h, wFf2T, ff2_b, x1f, (float*)d_out, nullptr, M, D, F,
        GF_RES | GF_WF32);
}
